// DCGRUCell_50354196579107
// MI455X (gfx1250) — matrix operand order repaired and checked by reading, not compiled, not run on hardware
//
#include <hip/hip_runtime.h>
#include <hip/hip_bf16.h>

typedef __attribute__((ext_vector_type(16))) __bf16        bf16x16;
typedef __attribute__((ext_vector_type(8)))  float          f32x8;
typedef __attribute__((ext_vector_type(8)))  unsigned short u16x8;
typedef __attribute__((ext_vector_type(16))) unsigned short u16x16;

#define DEV __device__ __forceinline__

// ---- bf16 helpers (round-to-nearest-even) ----
DEV unsigned short f2bf(float f) {
    unsigned int u = __builtin_bit_cast(unsigned int, f);
    u += 0x7FFFu + ((u >> 16) & 1u);
    return (unsigned short)(u >> 16);
}
DEV float bf2f(unsigned short s) {
    unsigned int u = ((unsigned int)s) << 16;
    return __builtin_bit_cast(float, u);
}
DEV bf16x16 make_frag(u16x8 lo, u16x8 hi) {
    u16x16 v = __builtin_shufflevector(lo, hi, 0,1,2,3,4,5,6,7,8,9,10,11,12,13,14,15);
    return __builtin_bit_cast(bf16x16, v);
}

// A fragment (16x32 bf16) from row-major matrix (row stride ld), K contiguous in memory.
// ISA: lane<16: row=row0+lane, K={0..7,16..23}; lane>=16: row=row0+lane-16, K={8..15,24..31}
DEV bf16x16 load_a_frag_g(const unsigned short* A, int row0, int ld, int k, int lane) {
    int r = row0 + (lane & 15);
    const unsigned short* p = A + (size_t)r * ld + k + ((lane >> 4) << 3);
    u16x8 lo = *(const u16x8*)p;
    u16x8 hi = *(const u16x8*)(p + 16);
    return make_frag(lo, hi);
}

// B fragment (32x16) from LDS transposed tile T[n][kk], row stride 40 ushorts.
// ISA: lane<16: col=ft+lane, K={0..7,16..23}; lane>=16: col=ft+lane-16, K={8..15,24..31}
DEV bf16x16 load_b_frag_s(const unsigned short* T, int ft, int lane) {
    const unsigned short* p = T + (size_t)(ft + (lane & 15)) * 40 + ((lane >> 4) << 3);
    u16x8 lo = *(const u16x8*)p;
    u16x8 hi = *(const u16x8*)(p + 16);
    return make_frag(lo, hi);
}

DEV f32x8 wmma_bf16(bf16x16 a, bf16x16 b, f32x8 c) {
    return __builtin_amdgcn_wmma_f32_16x16x32_bf16(false, a, false, b, (short)0, c, false, false);
}

// ---------------- support conversion + input packing (f-major stack) ----------------
__global__ __launch_bounds__(256) void cvt_supports(const float* __restrict__ s0,
                                                    const float* __restrict__ s1,
                                                    unsigned short* __restrict__ o0,
                                                    unsigned short* __restrict__ o1) {
    int i = blockIdx.x * 256 + threadIdx.x;  // exactly N*N threads
    o0[i] = f2bf(s0[i]);
    o1[i] = f2bf(s1[i]);
}

// stack layout: stkT[((b*5 + m)*128 + f)*1024 + n]   (n contiguous)
__global__ __launch_bounds__(256) void pack_xh(const float* __restrict__ x,
                                               const float* __restrict__ h,
                                               unsigned short* __restrict__ stkT) {
    size_t i = (size_t)blockIdx.x * 256 + threadIdx.x;  // B*128*1024 threads
    int n = (int)(i & 1023);
    int c = (int)((i >> 10) & 127);
    int b = (int)(i >> 17);
    float v = (c < 64) ? x[((size_t)b * 1024 + n) * 64 + c]
                       : h[((size_t)b * 1024 + n) * 64 + (c - 64)];
    stkT[((size_t)(b * 5) * 128 + c) * 1024 + n] = f2bf(v);
}

// ---------------- diffusion: Y[m_out] = alpha * S @ X[m_in]  (- X[m_prev]) -----------
// f-major stack => B fragments and D stores are contiguous: no LDS, no barriers.
// grid (64 batches, 8 row-tiles of 128); 8 waves: 2 row-tiles x 4 f-tiles each.
template <bool HAS_PREV>
__global__ __launch_bounds__(256) void diff_gemm(
    const unsigned short* __restrict__ S,      // 1024x1024 bf16, row-major
    const unsigned short* __restrict__ stkT,   // f-major stack
    unsigned short* __restrict__ stko,         // same base, slice m_out
    int m_in, int m_out, int m_prev, float alpha)
{
    const int b    = blockIdx.x;
    const int i0   = blockIdx.y * 128;
    const int tid  = threadIdx.x;
    const int wave = tid >> 5, lane = tid & 31;
    const int lh = lane >> 4, ll = lane & 15;
    const int rbase = i0 + (wave & 3) * 32;   // row tiles rbase, rbase+16
    const int fbase = (wave >> 2) * 64;       // f tiles fbase + 16t, t=0..3

    const unsigned short* Xin = stkT + (size_t)(b * 5 + m_in) * 128 * 1024;

    f32x8 acc[2][4] = {};
    for (int k = 0; k < 1024; k += 32) {
        int kp = (k + 32) & 1023;  // branchless prefetch address (wraps, always valid)
        __builtin_prefetch(S + (size_t)(rbase + ll) * 1024 + kp, 0, 3);
        __builtin_prefetch(Xin + (size_t)(fbase + ll) * 1024 + kp, 0, 3);
        bf16x16 af[2];
        #pragma unroll
        for (int rt = 0; rt < 2; ++rt)
            af[rt] = load_a_frag_g(S, rbase + rt * 16, 1024, k, lane);
        #pragma unroll
        for (int t = 0; t < 4; ++t) {
            const unsigned short* p =
                Xin + (size_t)(fbase + t * 16 + ll) * 1024 + k + (lh << 3);
            u16x8 lo = *(const u16x8*)p;
            u16x8 hi = *(const u16x8*)(p + 16);
            bf16x16 bfv = make_frag(lo, hi);
            #pragma unroll
            for (int rt = 0; rt < 2; ++rt)
                acc[rt][t] = wmma_bf16(af[rt], bfv, acc[rt][t]);
        }
    }

    unsigned short* Y = stko + (size_t)(b * 5 + m_out) * 128 * 1024;
    const unsigned short* P =
        HAS_PREV ? stkT + (size_t)(b * 5 + m_prev) * 128 * 1024 : nullptr;
    #pragma unroll
    for (int rt = 0; rt < 2; ++rt) {
        int rowb = rbase + rt * 16 + (lh << 3);  // 8 consecutive rows (contiguous n)
        #pragma unroll
        for (int t = 0; t < 4; ++t) {
            int f = fbase + t * 16 + ll;
            u16x8 o;
            if (HAS_PREV) {
                u16x8 pv = *(const u16x8*)(P + (size_t)f * 1024 + rowb);
                #pragma unroll
                for (int r = 0; r < 8; ++r)
                    o[r] = f2bf(alpha * acc[rt][t][r] - bf2f(pv[r]));
            } else {
                #pragma unroll
                for (int r = 0; r < 8; ++r)
                    o[r] = f2bf(alpha * acc[rt][t][r]);
            }
            *(u16x8*)(Y + (size_t)f * 1024 + rowb) = o;  // one b128 store per tile
        }
    }
}

// ---------------- projection: OUT = act(stack(65536x640) @ W(640xNCOL) + bias) -------
// A comes from the f-major stack via one LDS transpose stage per K-chunk.
// MODE 0: gates (sigmoid; write u, rewrite stack m0 cols 64..127 with bf16(r*h))
// MODE 1: candidate (tanh; h_new = u*h + (1-u)*c -> out)
template <int NCOL, int MODE>
__global__ __launch_bounds__(256) void proj_gemm(
    const unsigned short* __restrict__ stkT,
    const float* __restrict__ W,
    const float* __restrict__ bias,
    const float* __restrict__ h,
    float* __restrict__ ubuf,
    unsigned short* stk_m0,           // f-major base (gate epilogue writes)
    float* __restrict__ out)
{
    const int row0 = blockIdx.x * 128;   // global row = b*1024 + n ; 128 rows per block
    const int b    = row0 >> 10;
    const int n0   = row0 & 1023;
    const int tid  = threadIdx.x;
    const int wave = tid >> 5, lane = tid & 31;
    const int lh = lane >> 4, ll = lane & 15;
    constexpr int NT = NCOL / 16;
    __shared__ __align__(16) unsigned short AsT[128 * 40];   // [n][k] transposed
    __shared__ __align__(16) unsigned short WsT[NCOL * 40];  // [col][k] transposed

    f32x8 acc[NT] = {};
    for (int k = 0; k < 640; k += 32) {
        // stage A: 32 k-rows x 128 n (coalesced b128 reads), transpose into AsT
        for (int s = tid; s < 512; s += 256) {
            int j  = s >> 4;            // k index in chunk
            int no = (s & 15) << 3;     // n octet
            int kk = k + j;
            const unsigned short* src =
                stkT + ((size_t)(b * 5 + (kk >> 7)) * 128 + (kk & 127)) * 1024 + n0 + no;
            u16x8 v = *(const u16x8*)src;
            #pragma unroll
            for (int e = 0; e < 8; ++e) AsT[(no + e) * 40 + j] = v[e];
        }
        // stage W: 32 k x NCOL, fp32 -> bf16, transposed
        constexpr int OCT = NCOL / 8;
        for (int s = tid; s < 32 * OCT; s += 256) {
            int j  = s / OCT;
            int fo = (s % OCT) * 8;
            const float* src = W + (size_t)(k + j) * NCOL + fo;
            #pragma unroll
            for (int e = 0; e < 8; ++e) WsT[(fo + e) * 40 + j] = f2bf(src[e]);
        }
        __syncthreads();
        // A fragment: row n = wave*16 + ll, K chunks {0..7,16..23}/{8..15,24..31}
        const unsigned short* pa = AsT + (size_t)(wave * 16 + ll) * 40 + (lh << 3);
        bf16x16 af = make_frag(*(const u16x8*)pa, *(const u16x8*)(pa + 16));
        #pragma unroll
        for (int t = 0; t < NT; ++t) {
            bf16x16 bfv = load_b_frag_s(WsT, t * 16, lane);
            acc[t] = wmma_bf16(af, bfv, acc[t]);
        }
        __syncthreads();
    }

    const int rw = row0 + wave * 16 + (lh << 3);
    #pragma unroll
    for (int t = 0; t < NT; ++t) {
        #pragma unroll
        for (int r = 0; r < 8; ++r) {
            int row = rw + r;
            int col = t * 16 + ll;
            float v = acc[t][r] + bias[col];
            if (MODE == 0) {
                float sg = 1.f / (1.f + __expf(-v));
                if (col < 64) {  // r gate -> cand_in = [x | r*h]; x half already in place
                    float hv = h[(size_t)row * 64 + col];
                    int nn = n0 + (row - row0);
                    stk_m0[((size_t)(b * 5) * 128 + 64 + col) * 1024 + nn] = f2bf(sg * hv);
                } else {         // u gate
                    ubuf[(size_t)row * 64 + (col - 64)] = sg;
                }
            } else {
                float c  = tanhf(v);
                float u  = ubuf[(size_t)row * 64 + col];
                float hv = h[(size_t)row * 64 + col];
                out[(size_t)row * 64 + col] = u * hv + (1.f - u) * c;
            }
        }
    }
}

extern "C" void kernel_launch(void* const* d_in, const int* in_sizes, int n_in,
                              void* d_out, int out_size, void* d_ws, size_t ws_size,
                              hipStream_t stream) {
    const float* x  = (const float*)d_in[0];
    const float* h  = (const float*)d_in[1];
    const float* s0 = (const float*)d_in[2];
    const float* s1 = (const float*)d_in[3];
    const float* gw = (const float*)d_in[4];
    const float* gb = (const float*)d_in[5];
    const float* cw = (const float*)d_in[6];
    const float* cb = (const float*)d_in[7];
    float* out = (float*)d_out;

    // workspace layout
    unsigned short* S0b = (unsigned short*)d_ws;                 // 1024*1024 bf16
    unsigned short* S1b = S0b + (size_t)1024 * 1024;             // 1024*1024 bf16
    unsigned short* stkT = S1b + (size_t)1024 * 1024;            // 64*5*128*1024 bf16 (f-major)
    float* ubuf = (float*)(stkT + (size_t)64 * 5 * 128 * 1024);  // 64*1024*64 f32

    cvt_supports<<<4096, 256, 0, stream>>>(s0, s1, S0b, S1b);
    pack_xh<<<32768, 256, 0, stream>>>(x, h, stkT);

    dim3 dgrid(64, 8);
    // ---- gate pass: stack = [x0, S0 x0, 2 S0 S0 x0 - x0, S1 x0, 2 S1 S1 x0 - x0]
    diff_gemm<false><<<dgrid, 256, 0, stream>>>(S0b, stkT, stkT, 0, 1, -1, 1.f);
    diff_gemm<true ><<<dgrid, 256, 0, stream>>>(S0b, stkT, stkT, 1, 2,  0, 2.f);
    diff_gemm<false><<<dgrid, 256, 0, stream>>>(S1b, stkT, stkT, 0, 3, -1, 1.f);
    diff_gemm<true ><<<dgrid, 256, 0, stream>>>(S1b, stkT, stkT, 3, 4,  0, 2.f);
    proj_gemm<128, 0><<<512, 256, 0, stream>>>(stkT, gw, gb, h, ubuf, stkT, nullptr);
    // ---- candidate pass (stack m0 now holds [x | r*h])
    diff_gemm<false><<<dgrid, 256, 0, stream>>>(S0b, stkT, stkT, 0, 1, -1, 1.f);
    diff_gemm<true ><<<dgrid, 256, 0, stream>>>(S0b, stkT, stkT, 1, 2,  0, 2.f);
    diff_gemm<false><<<dgrid, 256, 0, stream>>>(S1b, stkT, stkT, 0, 3, -1, 1.f);
    diff_gemm<true ><<<dgrid, 256, 0, stream>>>(S1b, stkT, stkT, 3, 4,  0, 2.f);
    proj_gemm<64, 1><<<512, 256, 0, stream>>>(stkT, cw, cb, h, ubuf, nullptr, out);
}
